// CurrentLIFNetwork_39316130627640
// MI455X (gfx1250) — matrix operand order repaired and checked by reading, not compiled, not run on hardware
//
#include <hip/hip_runtime.h>

typedef __attribute__((ext_vector_type(16))) _Float16 v16h;
typedef __attribute__((ext_vector_type(8)))  _Float16 v8h;
typedef __attribute__((ext_vector_type(8)))  float    v8f;

// Physiological constants (all E/I params identical except beta and weight scale)
#define BETA_E 0.9512294245007140f   // exp(-1/20)
#define BETA_I 0.9048374180359596f   // exp(-1/10)
#define ALPHA  0.8187307530779818f   // exp(-1/5)
#define U_REST (-65.0f)
#define THETA  (-50.0f)
#define U_RESET (-65.0f)
#define R_MEM  100.0f

#define RED_STRIDE 12   // floats per lane slot in LDS reduce buffer (16B aligned, fewer bank conflicts)

// ---- Pack recurrent W (N x N, row-major, pre->post) into f16, column-major per
// ---- post-neuron n (so each WMMA B-fragment lane reads 16 contiguous halves),
// ---- folding in the E/I post-synaptic scaling.
__global__ void pack_w_kernel(const float* __restrict__ W, const int* __restrict__ nt,
                              _Float16* __restrict__ Wp, int N) {
    size_t id = (size_t)blockIdx.x * blockDim.x + threadIdx.x;
    size_t total = (size_t)N * N;
    if (id >= total) return;
    int n = (int)(id % N);       // post column (consecutive threads -> coalesced read)
    int k = (int)(id / N);       // pre row
    float scale = (nt[n] == 1) ? 0.5f : 2.0f;
    Wp[(size_t)n * N + k] = (_Float16)(W[(size_t)k * N + n] * scale);
}

// ---- Pack feedforward W (NIN x N) the same way (no scaling).
__global__ void pack_wff_kernel(const float* __restrict__ W, _Float16* __restrict__ Wp,
                                int N, int NIN) {
    size_t id = (size_t)blockIdx.x * blockDim.x + threadIdx.x;
    size_t total = (size_t)NIN * N;
    if (id >= total) return;
    int n = (int)(id % N);
    int k = (int)(id / N);
    Wp[(size_t)n * NIN + k] = (_Float16)W[(size_t)k * N + n];
}

// ---- Convert binary input spikes to f16 in [t][16][NIN] layout (rows 8..15 are
// ---- zero padding for the M=16 WMMA tile; zeroed by memset beforehand).
__global__ void pack_in_kernel(const float* __restrict__ in, _Float16* __restrict__ inh,
                               int T, int NIN) {
    size_t id = (size_t)blockIdx.x * blockDim.x + threadIdx.x;
    size_t total = (size_t)T * 8 * NIN;
    if (id >= total) return;
    int k = (int)(id % NIN);
    size_t tb = id / NIN;
    int b = (int)(tb % 8);
    int t = (int)(tb / 8);
    inh[((size_t)t * 16 + b) * NIN + k] =
        (_Float16)in[((size_t)b * T + t) * NIN + k];
}

// ---- Step 1 prologue: from (v0, I0) compute s_1, v_1; seed states + t=0 outputs.
__global__ void lif_init_kernel(const float* __restrict__ v0, const float* __restrict__ I0,
                                const int* __restrict__ nt,
                                float* __restrict__ out_s, float* __restrict__ out_v,
                                _Float16* __restrict__ s1h,      // spike buffer parity 1
                                float* __restrict__ v_state, float* __restrict__ I_state,
                                int N, int T) {
    int id = blockIdx.x * blockDim.x + threadIdx.x;
    if (id >= 8 * N) return;
    int n = id % N;
    int b = id / N;
    float beta = (nt[n] == 1) ? BETA_E : BETA_I;
    float I = I0[id];
    float vp = U_REST + (v0[id] - U_REST) * beta + I * (R_MEM * (1.0f - beta));
    float s = (vp >= THETA) ? 1.0f : 0.0f;
    float v = (s > 0.0f) ? U_RESET : vp;
    out_s[((size_t)b * T) * N + n] = s;
    out_v[((size_t)b * T) * N + n] = v;
    s1h[(size_t)b * N + n] = (_Float16)s;   // rows 8..15 stay zero (memset)
    v_state[id] = v;                        // v_1
    I_state[id] = I;                        // I_0
}

// ---- Per-timestep kernel. One block (8 waves) owns a 16-column tile of N.
// ---- K is split across the 8 waves (dependent-WMMA chain cut 128 -> 16, two
// ---- accumulators cut it to 8); partials are reduced through LDS. Then the
// ---- block advances I/v/s for its columns (column-local, no cross-block dep).
__global__ void __launch_bounds__(256)
lif_step_kernel(const _Float16* __restrict__ Wp,
                const _Float16* __restrict__ Wffp,
                const _Float16* __restrict__ inh,
                const _Float16* __restrict__ s_rd,   // s_t   [16][N], rows 8..15 zero
                _Float16* __restrict__ s_wr,         // s_t+1 [16][N]
                float* __restrict__ v_state,
                float* __restrict__ I_state,
                const int* __restrict__ nt,
                float* __restrict__ out_s, float* __restrict__ out_v,
                float* __restrict__ out_I,
                int N, int NIN, int T, int t) {
    __shared__ float redbuf[8 * 32 * RED_STRIDE];   // [wave][lane][padded v8f] = 12 KB

    const int tid    = threadIdx.x;
    const int lane   = tid & 31;
    const int wave   = tid >> 5;
    const int laneLo = lane & 15;
    const int laneHi = lane >> 4;
    const int n0     = blockIdx.x * 16;
    const int n      = n0 + laneLo;          // this lane's column

    const int KB   = N >> 5;                 // k-blocks of 32 (recurrent)
    const int KBW  = KB >> 3;                // per-wave k-blocks (16 for N=4096)
    const int kbBeg = wave * KBW;

    // A-fragment (16-bit A, 16x32): lanes 0-15 hold K 0..15 of row M=lane,
    // lanes 16-31 hold K 16..31 of row M=lane-16 -> two contiguous 16B loads.
    const _Float16* ap = s_rd + (size_t)laneLo * N + kbBeg * 32 + laneHi * 16;
    // B-fragment (32x16): lane = column laneLo, K-half laneHi*16 .. +16 contiguous.
    const _Float16* bp = Wp + (size_t)n * N + (size_t)kbBeg * 32 + laneHi * 16;

    // Scalar loop control (i, KBW are wave-invariant scalars); pointers carry the
    // per-wave offset and advance by a constant -> no per-iteration EXEC masking.
    v8f c0 = {}, c1 = {};                    // dual accumulators: chain 16 -> 8
    #pragma unroll 4
    for (int i = 0; i < KBW; i += 2) {
        {
            v8h alo = *(const v8h*)(ap);
            v8h ahi = *(const v8h*)(ap + 8);
            v16h a  = __builtin_shufflevector(alo, ahi,
                       0,1,2,3,4,5,6,7,8,9,10,11,12,13,14,15);
            v16h bm = *(const v16h*)(bp);
            c0 = __builtin_amdgcn_wmma_f32_16x16x32_f16(false, a, false, bm,
                                                        (short)0, c0, false, false);
        }
        {
            v8h alo = *(const v8h*)(ap + 32);
            v8h ahi = *(const v8h*)(ap + 40);
            v16h a  = __builtin_shufflevector(alo, ahi,
                       0,1,2,3,4,5,6,7,8,9,10,11,12,13,14,15);
            v16h bm = *(const v16h*)(bp + 32);
            c1 = __builtin_amdgcn_wmma_f32_16x16x32_f16(false, a, false, bm,
                                                        (short)0, c1, false, false);
        }
        ap += 64;
        bp += 64;
    }

    // Fused feedforward drive for this step: inputs[:, t-1, :] @ Wff.
    // KBF/8 k-blocks per wave (2 for NIN=512), alternating accumulators.
    const int KBF   = NIN >> 5;
    const int KBFW  = KBF >> 3;
    const int kfBeg = wave * KBFW;
    const _Float16* fap = inh + ((size_t)(t - 1) * 16 + laneLo) * NIN
                        + kfBeg * 32 + laneHi * 16;
    const _Float16* fbp = Wffp + (size_t)n * NIN + (size_t)kfBeg * 32 + laneHi * 16;
    #pragma unroll 2
    for (int i = 0; i < KBFW; ++i) {
        v8h alo = *(const v8h*)(fap);
        v8h ahi = *(const v8h*)(fap + 8);
        v16h a  = __builtin_shufflevector(alo, ahi,
                   0,1,2,3,4,5,6,7,8,9,10,11,12,13,14,15);
        v16h bm = *(const v16h*)(fbp);
        if (i & 1)
            c1 = __builtin_amdgcn_wmma_f32_16x16x32_f16(false, a, false, bm,
                                                        (short)0, c1, false, false);
        else
            c0 = __builtin_amdgcn_wmma_f32_16x16x32_f16(false, a, false, bm,
                                                        (short)0, c0, false, false);
        fap += 32;
        fbp += 32;
    }

    // Combine dual accumulators and park this wave's partial in LDS.
    v8f c = c0 + c1;
    *(v8f*)&redbuf[(size_t)(wave * 32 + lane) * RED_STRIDE] = c;  // 16B aligned -> ds_store_b128 x2
    __syncthreads();

    // Reduce 8 wave-partials + epilogue. C/D layout: VGPR i, lanes 0-15 ->
    // (M=i, N=lane); rows 0..7 = batches. Thread tid<128 owns (b, col).
    if (tid < 128) {
        const int col = tid & 15;
        const int b   = tid >> 4;
        const int nn  = n0 + col;
        float acc = 0.0f;
        #pragma unroll
        for (int w = 0; w < 8; ++w)
            acc += redbuf[(size_t)(w * 32 + col) * RED_STRIDE + b];

        const size_t sidx = (size_t)b * N + nn;
        const float Inew = ALPHA * I_state[sidx] + acc;
        I_state[sidx] = Inew;
        out_I[((size_t)b * T + (t - 1)) * N + nn] = Inew;

        if (t < T) {
            const float beta = (nt[nn] == 1) ? BETA_E : BETA_I;
            const float vprev = v_state[sidx];
            const float vp = U_REST + (vprev - U_REST) * beta
                           + Inew * (R_MEM * (1.0f - beta));
            const bool  sp = (vp >= THETA);
            const float vn = sp ? U_RESET : vp;
            v_state[sidx] = vn;
            s_wr[sidx] = (_Float16)(sp ? 1.0f : 0.0f);
            out_s[((size_t)b * T + t) * N + nn] = sp ? 1.0f : 0.0f;
            out_v[((size_t)b * T + t) * N + nn] = vn;
        }
    }
}

extern "C" void kernel_launch(void* const* d_in, const int* in_sizes, int n_in,
                              void* d_out, int out_size, void* d_ws, size_t ws_size,
                              hipStream_t stream) {
    const int*   nt     = (const int*)d_in[0];
    const float* Wrec   = (const float*)d_in[1];
    const float* Wff    = (const float*)d_in[2];
    const float* inputs = (const float*)d_in[3];
    const float* v0     = (const float*)d_in[4];
    const float* I0     = (const float*)d_in[5];

    const int N   = in_sizes[0];
    const int NIN = in_sizes[2] / N;
    const int B   = 8;
    const int T   = in_sizes[3] / (B * NIN);

    float* out_s = (float*)d_out;
    float* out_v = out_s + (size_t)B * T * N;
    float* out_I = out_v + (size_t)B * T * N;

    // Workspace carve-up (~53 MB for the reference sizes)
    char* ws = (char*)d_ws;
    auto alignup = [](size_t x) { return (x + 255) & ~(size_t)255; };
    size_t off = 0;
    _Float16* Wp   = (_Float16*)(ws + off); off = alignup(off + (size_t)N * N * 2);
    _Float16* Wffp = (_Float16*)(ws + off); off = alignup(off + (size_t)NIN * N * 2);
    _Float16* inh  = (_Float16*)(ws + off);
    size_t inh_bytes = (size_t)T * 16 * NIN * 2;           off = alignup(off + inh_bytes);
    _Float16* sbuf = (_Float16*)(ws + off);
    size_t s_bytes = (size_t)2 * 16 * N * 2;               off = alignup(off + s_bytes);
    float* v_state = (float*)(ws + off);                   off = alignup(off + (size_t)8 * N * 4);
    float* I_state = (float*)(ws + off);                   off = alignup(off + (size_t)8 * N * 4);

    // Zero padding rows (8..15) of the M=16 WMMA A-tiles (capture-safe memsets)
    hipMemsetAsync(inh,  0, inh_bytes, stream);
    hipMemsetAsync(sbuf, 0, s_bytes,   stream);

    const int TPB = 256;
    size_t nw = (size_t)N * N;
    pack_w_kernel  <<<(unsigned)((nw + TPB - 1) / TPB), TPB, 0, stream>>>(Wrec, nt, Wp, N);
    size_t nf = (size_t)NIN * N;
    pack_wff_kernel<<<(unsigned)((nf + TPB - 1) / TPB), TPB, 0, stream>>>(Wff, Wffp, N, NIN);
    size_t ni = (size_t)T * 8 * NIN;
    pack_in_kernel <<<(unsigned)((ni + TPB - 1) / TPB), TPB, 0, stream>>>(inputs, inh, T, NIN);

    _Float16* sbuf0 = sbuf;                     // parity 0
    _Float16* sbuf1 = sbuf + (size_t)16 * N;    // parity 1 (read by t=1)
    lif_init_kernel<<<(8 * N + TPB - 1) / TPB, TPB, 0, stream>>>(
        v0, I0, nt, out_s, out_v, sbuf1, v_state, I_state, N, T);

    // Scan: one kernel per timestep; spikes ping-pong between the two buffers.
    const int blocks = N / 16;   // one 16-column tile per 8-wave block
    for (int t = 1; t <= T; ++t) {
        const _Float16* s_rd = (t & 1) ? sbuf1 : sbuf0;
        _Float16*       s_wr = (t & 1) ? sbuf0 : sbuf1;
        lif_step_kernel<<<blocks, 256, 0, stream>>>(
            Wp, Wffp, inh, s_rd, s_wr, v_state, I_state, nt,
            out_s, out_v, out_I, N, NIN, T, t);
    }
}
